// MultiHeadSelfAttention_47579647705680
// MI455X (gfx1250) — matrix operand order repaired and checked by reading, not compiled, not run on hardware
//
#include <hip/hip_runtime.h>
#include <hip/hip_bf16.h>

typedef __bf16 bf16_t;
typedef __attribute__((ext_vector_type(16))) __bf16 v16bf;
typedef __attribute__((ext_vector_type(8)))  __bf16 v8bf;
typedef __attribute__((ext_vector_type(4)))  __bf16 v4bf;
typedef __attribute__((ext_vector_type(8)))  float  v8f;
typedef __attribute__((ext_vector_type(4)))  unsigned int u32x4;
typedef __attribute__((ext_vector_type(8)))  int i32x8;
typedef __attribute__((ext_vector_type(4)))  int i32x4;

static constexpr int SEQ   = 2048;
static constexpr int BATCH = 4;
static constexpr int HID   = 512;
static constexpr int NH    = 8;
static constexpr int HD    = 64;
static constexpr int MTOT  = BATCH * SEQ;   // 8192 rows
static constexpr int LDSP  = 80;            // padded LDS row stride (elems), 160B (16B aligned)

// ---------------------------------------------------------------- WMMA helpers

__device__ __forceinline__ v8f wmma_bf16(v16bf a, v16bf b, v8f c) {
  // D = A(16x32) * B(32x16) + C, f32 accumulate
  return __builtin_amdgcn_wmma_f32_16x16x32_bf16(false, a, false, b, (short)0, c,
                                                 false, false);
}

__device__ __forceinline__ v16bf join8(v8bf lo, v8bf hi) {
  v16bf f;
#pragma unroll
  for (int i = 0; i < 8; ++i) { f[i] = lo[i]; f[i + 8] = hi[i]; }
  return f;
}

// A fragment 16x32 bf16: rows m..m+15, contraction k0..k0+31.
// src row-major with leading dim ld (contraction contiguous).
// Lane layout: row = lane&15 ; lanes 0-15 hold K = {k0..k0+7, k0+16..k0+23},
//              lanes 16-31 hold K = {k0+8..k0+15, k0+24..k0+31}.
__device__ __forceinline__ v16bf load_a(const bf16_t* src, int ld, int m, int k0, int lane) {
  const bf16_t* p = src + (m + (lane & 15)) * ld + k0 + 8 * (lane >> 4);
  return join8(*(const v8bf*)p, *(const v8bf*)(p + 16));
}

// B fragment 32x16 bf16: output cols n..n+15, contraction k0..k0+31.
// src holds B^T row-major: row = output column, leading dim ld.
// Lane layout: col = lane&15 ; lanes 0-15 hold K = {k0..k0+7, k0+16..k0+23},
//              lanes 16-31 hold K = {k0+8..k0+15, k0+24..k0+31}.
__device__ __forceinline__ v16bf load_b(const bf16_t* src, int ld, int n, int k0, int lane) {
  const bf16_t* p = src + (n + (lane & 15)) * ld + k0 + 8 * (lane >> 4);
  return join8(*(const v8bf*)p, *(const v8bf*)(p + 16));
}

// ---------------------------------------------------------------- TDM helper
// Async DMA of a 64x64 bf16 tile (row stride 64 elems in global) into LDS at
// byte offset ldsOff, with LDS padding: +32B after every 128B row => row
// stride LDSP=80 elems. D# built per CDNA5 ISA.
//   group0: count=1 | lds_addr | global_addr(57b) | type=2
//   group1: data_size=2B, pad_enable, pad_interval=4 (32 DW), pad_amount=7 (8 DW),
//           tensor_dim0=64, tensor_dim1=SEQ, tile 64x64, dim0_stride=64
// This toolchain exposes the 6-arg builtin (extra int32x8 group, zero-filled).
__device__ __forceinline__ void tdm_load_tile_64x64(const bf16_t* gsrc, unsigned ldsOff) {
  unsigned long long ga = (unsigned long long)(const void*)gsrc;
  u32x4 g0;
  g0[0] = 1u;                                            // count=1, is_restore=0
  g0[1] = ldsOff;                                        // lds_addr (bytes)
  g0[2] = (unsigned)(ga & 0xffffffffu);                  // global_addr[31:0]
  g0[3] = (unsigned)((ga >> 32) & 0x01ffffffu) | (2u << 30);  // [56:32] | type=2
  i32x8 g1;
  g1[0] = (int)((1u << 16) |        // data_size = 2 bytes
                (1u << 20) |        // pad_enable
                (4u << 22) |        // pad_interval: 32 DWORDs (128B)
                (7u << 25));        // pad_amount: 8 DWORDs (32B)
  g1[1] = (int)(64u << 16);         // tensor_dim0 = 64 (low half in [63:48])
  g1[2] = (int)((unsigned)SEQ << 16); // tensor_dim1[15:0] in [95:80]
  g1[3] = (int)(64u << 16);         // tile_dim0 = 64 in [127:112]
  g1[4] = 64;                       // tile_dim1 = 64 in [143:128]
  g1[5] = 64;                       // tensor_dim0_stride = 64
  g1[6] = 0;
  g1[7] = 0;
  i32x4 z4 = {0, 0, 0, 0};
  i32x8 z8 = {0, 0, 0, 0, 0, 0, 0, 0};
  __builtin_amdgcn_tensor_load_to_lds(g0, g1, z4, z4, z8, 0);
}

// ---------------------------------------------------------------- kernel 1: f32 -> bf16

__global__ void cvt_f32_bf16(const float* __restrict__ src, bf16_t* __restrict__ dst, int n) {
  int i = (blockIdx.x * blockDim.x + threadIdx.x) * 4;
  if (i >= n) return;
  float4 f = *(const float4*)(src + i);
  v4bf o;
  o[0] = (bf16_t)f.x; o[1] = (bf16_t)f.y; o[2] = (bf16_t)f.z; o[3] = (bf16_t)f.w;
  *(v4bf*)(dst + i) = o;
}

// ---------------------------------------------------------------- kernel 2: Q/K/V projections
// y = x @ W^T  (x: [MTOT,HID] bf16, W: [HID,HID] bf16 row-major [out,in]).
// Per wave: 64x64 tile (4x4 WMMA subtiles). Block = 4 waves -> 128x128 tile.
// Output stored head-major: [B, H, S, HD] bf16.

__global__ __launch_bounds__(128) void qkv_gemm(
    const bf16_t* __restrict__ xb,
    const bf16_t* __restrict__ wq, const bf16_t* __restrict__ wk,
    const bf16_t* __restrict__ wv,
    bf16_t* __restrict__ qd, bf16_t* __restrict__ kd, bf16_t* __restrict__ vd) {
  const int lane = threadIdx.x & 31;
  const int wave = threadIdx.x >> 5;
  const bf16_t* W   = (blockIdx.z == 0) ? wq : (blockIdx.z == 1) ? wk : wv;
  bf16_t*       dst = (blockIdx.z == 0) ? qd : (blockIdx.z == 1) ? kd : vd;

  const int m0 = blockIdx.x * 128 + (wave & 1) * 64;
  const int n0 = blockIdx.y * 128 + (wave >> 1) * 64;

  v8f acc[4][4] = {};
  for (int k0 = 0; k0 < HID; k0 += 32) {
    v16bf af[4], bfr[4];
#pragma unroll
    for (int mt = 0; mt < 4; ++mt) af[mt] = load_a(xb, HID, m0 + mt * 16, k0, lane);
#pragma unroll
    for (int nt = 0; nt < 4; ++nt) bfr[nt] = load_b(W, HID, n0 + nt * 16, k0, lane);
#pragma unroll
    for (int mt = 0; mt < 4; ++mt)
#pragma unroll
      for (int nt = 0; nt < 4; ++nt)
        acc[mt][nt] = wmma_bf16(af[mt], bfr[nt], acc[mt][nt]);
  }

  const int lh = lane >> 4, ln = lane & 15;
#pragma unroll
  for (int mt = 0; mt < 4; ++mt)
#pragma unroll
    for (int nt = 0; nt < 4; ++nt)
#pragma unroll
      for (int r = 0; r < 8; ++r) {
        int m = m0 + mt * 16 + lh * 8 + r;   // global row
        int n = n0 + nt * 16 + ln;           // global col
        int b = m >> 11, s = m & (SEQ - 1);
        int h = n >> 6,  d = n & (HD - 1);
        dst[(((size_t)(b * NH + h)) * SEQ + s) * HD + d] = (bf16_t)acc[mt][nt][r];
      }
}

// ---------------------------------------------------------------- kernel 3: flash attention
// Grid: (SEQ/256, B*H). Block 256 threads = 8 waves; each wave owns 32 query rows.
// Per KV step (64 keys): TDM stages K [64x64] into LDS (padded rows) while all
// threads stage V transposed; S = Q K^T via WMMA, online softmax in f32,
// P bounced through LDS (C-layout -> A-layout), O += P V via WMMA.

__global__ __launch_bounds__(256) void flash_attn(
    const bf16_t* __restrict__ q, const bf16_t* __restrict__ k,
    const bf16_t* __restrict__ v, bf16_t* __restrict__ o) {
  // single shared block so sK sits at LDS byte offset 0 (TDM lds_addr)
  __shared__ bf16_t smem[64 * LDSP + HD * LDSP + 8 * 32 * LDSP];
  bf16_t* sK  = smem;                 // [kv][d]   (stride LDSP via TDM padding)
  bf16_t* sVt = smem + 64 * LDSP;     // [d][kv]   (transposed V)
  bf16_t* sP  = smem + (64 + HD) * LDSP;  // per-wave P scratch [32 rows][64 cols]

  const int lane = threadIdx.x & 31;
  const int wave = threadIdx.x >> 5;
  const int bh   = blockIdx.y;           // b*NH + h
  const bf16_t* qh = q + (size_t)bh * SEQ * HD;
  const bf16_t* kh = k + (size_t)bh * SEQ * HD;
  const bf16_t* vh = v + (size_t)bh * SEQ * HD;
  const int qrow0 = blockIdx.x * 256 + wave * 32;
  const int lh = lane >> 4, ln = lane & 15;
  const float inv_scale = 0.125f;        // 1/sqrt(64)

  // Q fragments stay resident: 2 row-subtiles x 2 contraction chunks of 32
  v16bf qf[2][2];
#pragma unroll
  for (int mt = 0; mt < 2; ++mt)
#pragma unroll
    for (int dc = 0; dc < 2; ++dc)
      qf[mt][dc] = load_a(qh, HD, qrow0 + mt * 16, dc * 32, lane);

  v8f oacc[2][4] = {};
  float mrow[2][8], lrow[2][8];
#pragma unroll
  for (int mt = 0; mt < 2; ++mt)
#pragma unroll
    for (int r = 0; r < 8; ++r) { mrow[mt][r] = -1e30f; lrow[mt][r] = 0.f; }

  bf16_t* pw = sP + wave * 32 * LDSP;

  for (int kv0 = 0; kv0 < SEQ; kv0 += 64) {
    __syncthreads();   // previous iteration's LDS reads done before overwrite

    // ---- async TDM: K tile -> sK (LDS offset 0), rows padded to LDSP elems
    if (wave == 0) tdm_load_tile_64x64(kh + (size_t)kv0 * HD, 0u);

    // ---- stage V transposed (256 threads x 2 x 8 elems = 64x64), overlaps TDM
#pragma unroll
    for (int j = 0; j < 2; ++j) {
      int off = (threadIdx.x + j * 256) * 8;
      int row = off >> 6, d = off & 63;
      v8bf vv = *(const v8bf*)(vh + (size_t)(kv0 + row) * HD + d);
#pragma unroll
      for (int e = 0; e < 8; ++e) sVt[(d + e) * LDSP + row] = vv[e];
    }

    if (wave == 0) __builtin_amdgcn_s_wait_tensorcnt(0);
    __syncthreads();

    // ---- S = Q K^T  (2 row-subtiles x 4 key-subtiles)
    v8f sacc[2][4] = {};
#pragma unroll
    for (int nt = 0; nt < 4; ++nt)
#pragma unroll
      for (int dc = 0; dc < 2; ++dc) {
        v16bf bfr = load_b(sK, LDSP, nt * 16, dc * 32, lane);
#pragma unroll
        for (int mt = 0; mt < 2; ++mt)
          sacc[mt][nt] = wmma_bf16(qf[mt][dc], bfr, sacc[mt][nt]);
      }

    // ---- online softmax (row stats live per-lane for 8 rows per subtile)
#pragma unroll
    for (int mt = 0; mt < 2; ++mt)
#pragma unroll
      for (int r = 0; r < 8; ++r) {
        float mx = -1e30f;
#pragma unroll
        for (int nt = 0; nt < 4; ++nt) mx = fmaxf(mx, (float)sacc[mt][nt][r]);
#pragma unroll
        for (int off = 1; off < 16; off <<= 1) mx = fmaxf(mx, __shfl_xor(mx, off, 32));
        float mnew  = fmaxf(mrow[mt][r], mx * inv_scale);
        float alpha = __expf(mrow[mt][r] - mnew);
        mrow[mt][r] = mnew;
        float rsum = 0.f;
#pragma unroll
        for (int nt = 0; nt < 4; ++nt) {
          float pj = __expf((float)sacc[mt][nt][r] * inv_scale - mnew);
          sacc[mt][nt][r] = pj;
          rsum += pj;
        }
#pragma unroll
        for (int off = 1; off < 16; off <<= 1) rsum += __shfl_xor(rsum, off, 32);
        lrow[mt][r] = lrow[mt][r] * alpha + rsum;
#pragma unroll
        for (int nt = 0; nt < 4; ++nt)
          oacc[mt][nt][r] = oacc[mt][nt][r] * alpha;
      }

    // ---- write P (bf16) to per-wave LDS scratch to re-layout C-frag -> A-frag
#pragma unroll
    for (int mt = 0; mt < 2; ++mt)
#pragma unroll
      for (int nt = 0; nt < 4; ++nt)
#pragma unroll
        for (int r = 0; r < 8; ++r)
          pw[(mt * 16 + lh * 8 + r) * LDSP + nt * 16 + ln] = (bf16_t)sacc[mt][nt][r];

    v16bf pf[2][2];
#pragma unroll
    for (int mt = 0; mt < 2; ++mt)
#pragma unroll
      for (int kc = 0; kc < 2; ++kc)
        pf[mt][kc] = load_a(pw, LDSP, mt * 16, kc * 32, lane);

    // ---- O += P @ V   (V^T in LDS supplies B fragments: col = head dim)
#pragma unroll
    for (int nt = 0; nt < 4; ++nt)
#pragma unroll
      for (int kc = 0; kc < 2; ++kc) {
        v16bf bfr = load_b(sVt, LDSP, nt * 16, kc * 32, lane);
#pragma unroll
        for (int mt = 0; mt < 2; ++mt)
          oacc[mt][nt] = wmma_bf16(pf[mt][kc], bfr, oacc[mt][nt]);
      }
  }

  // ---- epilogue: normalize, write O as [B, S, H*HD] bf16 (row-major for final GEMM)
  const int b = bh >> 3, h = bh & (NH - 1);
#pragma unroll
  for (int mt = 0; mt < 2; ++mt)
#pragma unroll
    for (int r = 0; r < 8; ++r) {
      float rinv = 1.0f / lrow[mt][r];
      int srow = qrow0 + mt * 16 + lh * 8 + r;
#pragma unroll
      for (int nt = 0; nt < 4; ++nt) {
        int d = nt * 16 + ln;
        o[((size_t)(b * SEQ + srow)) * HID + h * HD + d] =
            (bf16_t)(oacc[mt][nt][r] * rinv);
      }
    }
}

// ---------------------------------------------------------------- kernel 4: output projection
// out = O @ Wo^T, f32 output to d_out [MTOT, HID].

__global__ __launch_bounds__(128) void out_gemm(
    const bf16_t* __restrict__ a, const bf16_t* __restrict__ w,
    float* __restrict__ out) {
  const int lane = threadIdx.x & 31;
  const int wave = threadIdx.x >> 5;
  const int m0 = blockIdx.x * 128 + (wave & 1) * 64;
  const int n0 = blockIdx.y * 128 + (wave >> 1) * 64;

  v8f acc[4][4] = {};
  for (int k0 = 0; k0 < HID; k0 += 32) {
    v16bf af[4], bfr[4];
#pragma unroll
    for (int mt = 0; mt < 4; ++mt) af[mt] = load_a(a, HID, m0 + mt * 16, k0, lane);
#pragma unroll
    for (int nt = 0; nt < 4; ++nt) bfr[nt] = load_b(w, HID, n0 + nt * 16, k0, lane);
#pragma unroll
    for (int mt = 0; mt < 4; ++mt)
#pragma unroll
      for (int nt = 0; nt < 4; ++nt)
        acc[mt][nt] = wmma_bf16(af[mt], bfr[nt], acc[mt][nt]);
  }

  const int lh = lane >> 4, ln = lane & 15;
#pragma unroll
  for (int mt = 0; mt < 4; ++mt)
#pragma unroll
    for (int nt = 0; nt < 4; ++nt)
#pragma unroll
      for (int r = 0; r < 8; ++r) {
        int m = m0 + mt * 16 + lh * 8 + r;
        int n = n0 + nt * 16 + ln;
        out[(size_t)m * HID + n] = acc[mt][nt][r];
      }
}

// ---------------------------------------------------------------- launch

extern "C" void kernel_launch(void* const* d_in, const int* in_sizes, int n_in,
                              void* d_out, int out_size, void* d_ws, size_t ws_size,
                              hipStream_t stream) {
  const float* x  = (const float*)d_in[0];
  const float* Wq = (const float*)d_in[1];
  const float* Wk = (const float*)d_in[2];
  const float* Wv = (const float*)d_in[3];
  const float* Wo = (const float*)d_in[4];
  float* out = (float*)d_out;

  // workspace layout (bf16), ~44 MB total
  bf16_t* xb  = (bf16_t*)d_ws;
  bf16_t* wqb = xb  + (size_t)MTOT * HID;
  bf16_t* wkb = wqb + (size_t)HID * HID;
  bf16_t* wvb = wkb + (size_t)HID * HID;
  bf16_t* wob = wvb + (size_t)HID * HID;
  bf16_t* qb  = wob + (size_t)HID * HID;
  bf16_t* kb  = qb  + (size_t)MTOT * HID;
  bf16_t* vb  = kb  + (size_t)MTOT * HID;
  bf16_t* ob  = vb  + (size_t)MTOT * HID;

  cvt_f32_bf16<<<(MTOT * HID / 4) / 256, 256, 0, stream>>>(x, xb, MTOT * HID);
  cvt_f32_bf16<<<(HID * HID / 4) / 256, 256, 0, stream>>>(Wq, wqb, HID * HID);
  cvt_f32_bf16<<<(HID * HID / 4) / 256, 256, 0, stream>>>(Wk, wkb, HID * HID);
  cvt_f32_bf16<<<(HID * HID / 4) / 256, 256, 0, stream>>>(Wv, wvb, HID * HID);
  cvt_f32_bf16<<<(HID * HID / 4) / 256, 256, 0, stream>>>(Wo, wob, HID * HID);

  qkv_gemm<<<dim3(MTOT / 128, HID / 128, 3), 128, 0, stream>>>(xb, wqb, wkb, wvb,
                                                               qb, kb, vb);
  flash_attn<<<dim3(SEQ / 256, BATCH * NH), 256, 0, stream>>>(qb, kb, vb, ob);
  out_gemm<<<dim3(MTOT / 128, HID / 128), 128, 0, stream>>>(ob, wob, out);
}
